// ProxyNCA_37555194036773
// MI455X (gfx1250) — matrix operand order repaired and checked by reading, not compiled, not run on hardware
//
#include <hip/hip_runtime.h>
#include <hip/hip_bf16.h>

// ---------------- problem dimensions (fixed by setup_inputs) ----------------
#define B_DIM   4096
#define D_DIM   512
#define C_DIM   1000
#define NB      6
#define PTOT    6000
#define PT_PAD  6016          // 47 * 128
#define CL_PAD  1024          // 8 * 128
#define TOPK    600

// ---------------- bf16 helpers ----------------
__device__ inline unsigned short f2bf(float f) {
  unsigned int u = __float_as_uint(f);
  unsigned int r = u + 0x7FFFu + ((u >> 16) & 1u);   // round-to-nearest-even
  return (unsigned short)(r >> 16);
}
__device__ inline float bf2f(unsigned short u) {
  return __uint_as_float(((unsigned int)u) << 16);
}
// monotonic float -> uint key (total order)
__device__ inline unsigned int fkey(float f) {
  unsigned int u = __float_as_uint(f);
  return (u & 0x80000000u) ? ~u : (u | 0x80000000u);
}

// ---------------- CDNA5 async global->LDS copy (16B per lane) ----------------
// GLOBAL_LOAD_ASYNC_TO_LDS_B128, GV mode: vdst = LDS byte address, vaddr = 64-bit
// global address. Tracked by ASYNCcnt (in-order completion among async loads).
__device__ inline void async_b128(const void* gsrc, void* ldst) {
  unsigned int lds = (unsigned int)(unsigned long long)ldst;   // low 32 bits = LDS offset
  unsigned long long gaddr = (unsigned long long)gsrc;
  asm volatile("global_load_async_to_lds_b128 %0, %1, off"
               :: "v"(lds), "v"(gaddr) : "memory");
}
__device__ inline void wait_async_0() {
  asm volatile("s_wait_asynccnt 0x0" ::: "memory");
}

// ---------------- WMMA types ----------------
typedef __attribute__((ext_vector_type(16))) __bf16 v16bf;
typedef __attribute__((ext_vector_type(8)))  float  v8f;
union FragB16 { unsigned int u[8]; v16bf v; };

// ---------------- init accumulators ----------------
__global__ void init_acc_kernel(float* acc) {
  if (threadIdx.x < 2) acc[threadIdx.x] = 0.0f;
}

// ---------------- normalize X rows -> bf16 Xn[4096][512] ----------------
__global__ __launch_bounds__(256) void norm_x_kernel(const float* __restrict__ X,
                                                     unsigned short* __restrict__ Xn) {
  __shared__ float red[256];
  int b = blockIdx.x, tid = threadIdx.x;
  const float* row = X + (size_t)b * D_DIM;
  float a0 = row[tid], a1 = row[tid + 256];
  red[tid] = a0 * a0 + a1 * a1;
  __syncthreads();
  for (int off = 128; off > 0; off >>= 1) {
    if (tid < off) red[tid] += red[tid + off];
    __syncthreads();
  }
  float scale = 1.0f / fmaxf(sqrtf(red[0]), 1e-12f);
  unsigned short* out = Xn + (size_t)b * D_DIM;
  out[tid]       = f2bf(a0 * scale);
  out[tid + 256] = f2bf(a1 * scale);
}

// ------- normalize proxy columns -> Pt[6016][512] bf16 (proxy-major, padded) -------
__global__ __launch_bounds__(256) void norm_p_kernel(const float* __restrict__ proxies,
                                                     unsigned short* __restrict__ Pt) {
  int p = blockIdx.x * 256 + threadIdx.x;
  if (p >= PT_PAD) return;
  if (p >= PTOT) {  // zero padding rows
    for (int d = 0; d < D_DIM; ++d) Pt[(size_t)p * D_DIM + d] = 0;
    return;
  }
  float s = 0.0f;
  for (int d = 0; d < D_DIM; ++d) {
    float v = proxies[(size_t)d * PTOT + p];
    s += v * v;
  }
  float scale = 1.0f / fmaxf(sqrtf(s), 1e-12f);
  for (int d = 0; d < D_DIM; ++d) {
    float v = proxies[(size_t)d * PTOT + p] * scale;
    Pt[(size_t)p * D_DIM + d] = f2bf(v);
  }
}

// ------- PsumT[c][k] = sum over the 6 proxies of class c of Pt[6c+j][k] -------
__global__ __launch_bounds__(256) void psum_kernel(const unsigned short* __restrict__ Pt,
                                                   unsigned short* __restrict__ PsumT) {
  int i = blockIdx.x * 256 + threadIdx.x;
  if (i >= CL_PAD * D_DIM) return;
  int c = i >> 9, k = i & (D_DIM - 1);
  float s = 0.0f;
  if (c < C_DIM) {
    #pragma unroll
    for (int j = 0; j < NB; ++j) s += bf2f(Pt[(size_t)(c * NB + j) * D_DIM + k]);
  }
  PsumT[i] = f2bf(s);
}

// ---------------- tiled bf16 WMMA GEMM: C[M][N] = A[M][K] * Bt[N][K]^T ----------------
// A row-major (lda), Bt N-major (ldbt), C row-major f32 (ldc).
// Block tile 128x128, 8 waves (2 along M x 4 along N), each wave 64x32, K-step 32.
// Double-buffered LDS staging via async global->LDS loads (ASYNCcnt pipeline):
//   wait(tile k) -> barrier -> issue tile k+1 (async) -> WMMA on tile k
__global__ __launch_bounds__(256) void gemm_bf16_kernel(const __bf16* __restrict__ A,
                                                        const __bf16* __restrict__ Bt,
                                                        float* __restrict__ C,
                                                        int K, int lda, int ldbt, int ldc) {
  __shared__ __align__(16) unsigned short As[2][128][32];  // [buf][m][k]
  __shared__ __align__(16) unsigned short Bs[2][128][32];  // [buf][n][k]

  int tid  = threadIdx.x;
  int lane = tid & 31;
  int wave = tid >> 5;
  int wm = wave & 1;        // 0..1  -> 64 rows each
  int wn = wave >> 1;       // 0..3  -> 32 cols each
  int bm = blockIdx.y * 128;
  int bn = blockIdx.x * 128;
  int half = lane >> 4;     // K-half selector
  int lc   = lane & 15;     // row (A) / col (B,C) within 16

  // straight-line staging: thread covers rows r and r+64 at bf16 column c
  int sr = tid >> 2;             // 0..63
  int sc = (tid & 3) << 3;       // 0, 8, 16, 24 (bf16 elements)

  v8f acc[4][2];
  #pragma unroll
  for (int i = 0; i < 4; ++i)
    #pragma unroll
    for (int j = 0; j < 2; ++j)
      acc[i][j] = (v8f){0.f,0.f,0.f,0.f,0.f,0.f,0.f,0.f};

  // async-stage one 128x32 A tile + one 128x32 Bt tile into buffer `buf`
  auto stage = [&](int buf, int k0) {
    async_b128(A  + (size_t)(bm + sr)      * lda  + k0 + sc, &As[buf][sr][sc]);
    async_b128(A  + (size_t)(bm + sr + 64) * lda  + k0 + sc, &As[buf][sr + 64][sc]);
    async_b128(Bt + (size_t)(bn + sr)      * ldbt + k0 + sc, &Bs[buf][sr][sc]);
    async_b128(Bt + (size_t)(bn + sr + 64) * ldbt + k0 + sc, &Bs[buf][sr + 64][sc]);
  };

  int nIter = K >> 5;
  stage(0, 0);
  for (int it = 0; it < nIter; ++it) {
    int buf = it & 1;
    wait_async_0();                    // this wave's current tile is in LDS
    __syncthreads();                   // all waves' tile parts visible; prev compute done
    if (it + 1 < nIter)
      stage(buf ^ 1, (it + 1) << 5);   // prefetch next tile while computing this one

    // B fragments (contiguous k order used here: lanes 0-15 K=0-15, lanes 16-31 K=16-31)
    FragB16 bfrag[2];
    #pragma unroll
    for (int jn = 0; jn < 2; ++jn) {
      int nb = wn * 32 + jn * 16 + lc;
      #pragma unroll
      for (int j = 0; j < 8; ++j) {
        int kk = half * 16 + j * 2;
        bfrag[jn].u[j] = *reinterpret_cast<const unsigned int*>(&Bs[buf][nb][kk]);
      }
    }
    #pragma unroll
    for (int im = 0; im < 4; ++im) {
      // A fragment (contiguous k order, same as B): VGPR j -> K = half*16 + 2j (+0/1)
      FragB16 afrag;
      int mb = wm * 64 + im * 16 + lc;
      #pragma unroll
      for (int j = 0; j < 8; ++j) {
        int kk = half * 16 + j * 2;
        afrag.u[j] = *reinterpret_cast<const unsigned int*>(&As[buf][mb][kk]);
      }
      #pragma unroll
      for (int jn = 0; jn < 2; ++jn) {
        acc[im][jn] = __builtin_amdgcn_wmma_f32_16x16x32_bf16(
            false, afrag.v, false, bfrag[jn].v, (short)0, acc[im][jn], false, false);
      }
    }
  }

  // store C (f32 16x16 C/D layout: VGPR e -> M = half*8 + e, N = lane&15)
  #pragma unroll
  for (int im = 0; im < 4; ++im) {
    #pragma unroll
    for (int jn = 0; jn < 2; ++jn) {
      int n = bn + wn * 32 + jn * 16 + lc;
      #pragma unroll
      for (int e = 0; e < 8; ++e) {
        int m = bm + wm * 64 + im * 16 + half * 8 + e;
        C[(size_t)m * ldc + n] = acc[im][jn][e];
      }
    }
  }
}

// ---------------- classify loss: per-row top-600 mask + masked softmax CE ----------
__global__ __launch_bounds__(256) void classify_kernel(const float* __restrict__ sim,
                                                       const int* __restrict__ T,
                                                       float* __restrict__ acc) {
  __shared__ float svals[PTOT];
  __shared__ unsigned int hist[256];
  __shared__ unsigned int s_prefix, s_k;
  __shared__ float s_red[256];
  __shared__ float s_target;

  int b = blockIdx.x, tid = threadIdx.x;
  int Tb = T[b];
  const float* row = sim + (size_t)b * PT_PAD;
  for (int p = tid; p < PTOT; p += 256) svals[p] = row[p];
  if (tid == 0) { s_prefix = 0u; s_k = TOPK; s_target = 0.0f; }
  __syncthreads();

  int pLo = NB * Tb, pHi = NB * Tb + NB;

  // 4-pass radix select: exact key of the 600th-largest boosted value
  for (int pass = 0; pass < 4; ++pass) {
    int shift = 24 - pass * 8;
    hist[tid] = 0u;
    __syncthreads();
    unsigned int pref = s_prefix;
    unsigned int prefmask = (pass == 0) ? 0u : (0xFFFFFFFFu << (shift + 8));
    for (int p = tid; p < PTOT; p += 256) {
      float v = svals[p] + ((p >= pLo && p < pHi) ? 1000.0f : 0.0f);
      unsigned int key = fkey(v);
      if ((key & prefmask) == pref) atomicAdd(&hist[(key >> shift) & 255u], 1u);
    }
    __syncthreads();
    if (tid == 0) {
      unsigned int k = s_k, cum = 0u;
      int d = 255;
      for (; d > 0; --d) {
        if (cum + hist[d] >= k) break;
        cum += hist[d];
      }
      s_prefix = pref | (((unsigned int)d) << shift);
      s_k = k - cum;
    }
    __syncthreads();
  }
  unsigned int tau = s_prefix;

  // per-class masked sums -> masked softmax denominator
  float sum_e = 0.0f;
  for (int c = tid; c < C_DIM; c += 256) {
    float boost = (c == Tb) ? 1000.0f : 0.0f;
    float s = 0.0f;
    bool any = false;
    #pragma unroll
    for (int j = 0; j < NB; ++j) {
      float v = svals[c * NB + j];
      if (fkey(v + boost) >= tau) { s += v; any = true; }
    }
    if (any) sum_e += __expf(s);
    if (c == Tb) s_target = s;   // all positives are selected (boost >> 1)
  }
  s_red[tid] = sum_e;
  __syncthreads();
  for (int off = 128; off > 0; off >>= 1) {
    if (tid < off) s_red[tid] += s_red[tid + off];
    __syncthreads();
  }
  if (tid == 0) {
    float predict = __expf(s_target) / (1e-8f + s_red[0]);
    atomicAdd(&acc[0], -__logf(predict + 1e-20f));
  }
}

// ---------------- proxy regularizer: row-wise CE over centers_logits ----------------
__global__ __launch_bounds__(256) void reg_kernel(const float* __restrict__ CL,
                                                  float* __restrict__ acc) {
  __shared__ float red[256];
  int p = blockIdx.x, tid = threadIdx.x;
  const float* row = CL + (size_t)p * CL_PAD;
  float m = -3.4e38f;
  for (int c = tid; c < C_DIM; c += 256) m = fmaxf(m, row[c]);
  red[tid] = m;
  __syncthreads();
  for (int off = 128; off > 0; off >>= 1) {
    if (tid < off) red[tid] = fmaxf(red[tid], red[tid + off]);
    __syncthreads();
  }
  float mx = red[0];
  __syncthreads();
  float se = 0.0f;
  for (int c = tid; c < C_DIM; c += 256) se += __expf(row[c] - mx);
  red[tid] = se;
  __syncthreads();
  for (int off = 128; off > 0; off >>= 1) {
    if (tid < off) red[tid] += red[tid + off];
    __syncthreads();
  }
  if (tid == 0) {
    float lse = mx + __logf(red[0]);
    atomicAdd(&acc[1], lse - row[p / NB]);
  }
}

// ---------------- finalize ----------------
__global__ void finalize_kernel(const float* __restrict__ acc, float* __restrict__ out) {
  if (threadIdx.x == 0)
    out[0] = acc[0] / (float)B_DIM + 0.3f * (acc[1] / (float)PTOT);
}

// ---------------- launcher ----------------
extern "C" void kernel_launch(void* const* d_in, const int* in_sizes, int n_in,
                              void* d_out, int out_size, void* d_ws, size_t ws_size,
                              hipStream_t stream) {
  const float* X       = (const float*)d_in[0];
  const float* proxies = (const float*)d_in[1];
  const int*   T       = (const int*)d_in[2];
  float* out = (float*)d_out;

  char* ws = (char*)d_ws;
  const size_t ACC_OFF   = 0;
  const size_t XN_OFF    = 256;
  const size_t PT_OFF    = XN_OFF   + (size_t)B_DIM * D_DIM * 2;    // 4 MB
  const size_t PSUMT_OFF = PT_OFF   + (size_t)PT_PAD * D_DIM * 2;   // 6.2 MB
  const size_t SIM_OFF   = PSUMT_OFF + (size_t)CL_PAD * D_DIM * 2;  // 1 MB
  // sim: 4096x6016 f32 (98.6 MB); CL (6016x1024 f32) aliases sim (stream-ordered)

  float*          acc   = (float*)(ws + ACC_OFF);
  unsigned short* Xn    = (unsigned short*)(ws + XN_OFF);
  unsigned short* Pt    = (unsigned short*)(ws + PT_OFF);
  unsigned short* PsumT = (unsigned short*)(ws + PSUMT_OFF);
  float*          sim   = (float*)(ws + SIM_OFF);
  float*          CL    = (float*)(ws + SIM_OFF);

  init_acc_kernel<<<1, 32, 0, stream>>>(acc);
  norm_x_kernel<<<B_DIM, 256, 0, stream>>>(X, Xn);
  norm_p_kernel<<<(PT_PAD + 255) / 256, 256, 0, stream>>>(proxies, Pt);
  psum_kernel<<<(CL_PAD * D_DIM + 255) / 256, 256, 0, stream>>>(Pt, PsumT);

  // sim = Xn (4096x512) * Pt^T (512x6016)
  gemm_bf16_kernel<<<dim3(PT_PAD / 128, B_DIM / 128), 256, 0, stream>>>(
      (const __bf16*)Xn, (const __bf16*)Pt, sim, D_DIM, D_DIM, D_DIM, PT_PAD);

  classify_kernel<<<B_DIM, 256, 0, stream>>>(sim, T, acc);

  // centers_logits = Pt (6016x512) * PsumT^T (512x1024)
  gemm_bf16_kernel<<<dim3(CL_PAD / 128, PT_PAD / 128), 256, 0, stream>>>(
      (const __bf16*)Pt, (const __bf16*)PsumT, CL, D_DIM, D_DIM, D_DIM, CL_PAD);

  reg_kernel<<<PTOT, 256, 0, stream>>>(CL, acc);
  finalize_kernel<<<1, 32, 0, stream>>>(acc, out);
}
